// Transformer_Block_1417339207765
// MI455X (gfx1250) — matrix operand order repaired and checked by reading, not compiled, not run on hardware
//
#include <hip/hip_runtime.h>

// ---------------------------------------------------------------------------
// MI455X (gfx1250) transformer decoder block.
// All matrix math via V_WMMA_F32_16X16X32_F16 (wave32, f32 accumulate).
// Every GEMM here is 1024x1024x1024 per batch (E == S == 1024), batch = 8.
// GEMM: block tile 128x256, 8 waves (2x4), wave tile 64x64 -> 16 WMMA per
// 16 b128 loads per k-step of 32 (1.0 load/WMMA issue ratio).
// ---------------------------------------------------------------------------

typedef __attribute__((ext_vector_type(16))) _Float16 v16h;
typedef __attribute__((ext_vector_type(8)))  _Float16 v8h;
typedef __attribute__((ext_vector_type(8)))  float    v8f;

constexpr int kL = 1024;   // E == S == 1024
constexpr int kB = 8;      // batch

// ---------------- block-wide reductions (wave32 -> 8 waves via LDS) --------
__device__ __forceinline__ float block_sum(float v, float* sbuf) {
  #pragma unroll
  for (int o = 16; o > 0; o >>= 1) v += __shfl_xor(v, o, 32);
  __syncthreads();
  if ((threadIdx.x & 31) == 0) sbuf[threadIdx.x >> 5] = v;
  __syncthreads();
  float t = 0.f;
  #pragma unroll
  for (int i = 0; i < 8; ++i) t += sbuf[i];
  return t;
}

__device__ __forceinline__ float block_max(float v, float* sbuf) {
  #pragma unroll
  for (int o = 16; o > 0; o >>= 1) v = fmaxf(v, __shfl_xor(v, o, 32));
  __syncthreads();
  if ((threadIdx.x & 31) == 0) sbuf[threadIdx.x >> 5] = v;
  __syncthreads();
  float t = sbuf[0];
  #pragma unroll
  for (int i = 1; i < 8; ++i) t = fmaxf(t, sbuf[i]);
  return t;
}

// ---------------- f32 -> f16 convert (4 elems/thread) ----------------------
__global__ __launch_bounds__(256) void f32_to_f16_kernel(
    const float* __restrict__ src, _Float16* __restrict__ dst) {
  long i0 = (long)blockIdx.x * 1024 + threadIdx.x;
  #pragma unroll
  for (int i = 0; i < 4; ++i) dst[i0 + i * 256] = (_Float16)src[i0 + i * 256];
}

// ---------------- weight transpose + convert: WT[o][k] = (h)W[k][o] --------
__global__ __launch_bounds__(256) void transpose_convert_kernel(
    const float* __restrict__ W, _Float16* __restrict__ WT) {
  __shared__ float tile[32][33];
  const int bx = blockIdx.x * 32;   // o
  const int by = blockIdx.y * 32;   // k
  const int tx = threadIdx.x, ty = threadIdx.y;   // block (32,8)
  #pragma unroll
  for (int i = 0; i < 4; ++i)
    tile[ty + i * 8][tx] = W[(long)(by + ty + i * 8) * kL + bx + tx];
  __syncthreads();
  #pragma unroll
  for (int i = 0; i < 4; ++i)
    WT[(long)(bx + ty + i * 8) * kL + by + tx] = (_Float16)tile[tx][ty + i * 8];
}

// ---------------------------------------------------------------------------
// Batched GEMM:  C[b][m][n] = scale * sum_k A[b][m][k] * BT[b][n][k]  (+bias)
// A, BT f16 row-major (ld = 1024).
// cmode: 0 = full; 1 = causal scores (skip blocks fully above the diagonal);
//        2 = causal prob@V (K-loop clamped to this row-block's diagonal).
// Fragment layouts per CDNA5 ISA 7.12.2:
//   A 16x32 f16: lane<16 -> row m, K {0..7}U{16..23}; lane>=16 -> K {8..15}U{24..31}
//   B consumed from BT[N,K]: lane<16 -> col n, K {0..7}U{16..23}; lane>=16 -> K {8..15}U{24..31}
//   C f32: lane<16 vgpr j -> (m0+j, n); lane>=16 -> (m0+8+j, n)
// TRANS_OUT writes f16 C^T (8 contiguous halves per lane -> one b128 store).
// ---------------------------------------------------------------------------
template <bool OUT_HALF, bool TRANS_OUT, bool HAS_BIAS, bool DO_RELU>
__global__ __launch_bounds__(256) void gemm_kernel(
    const _Float16* __restrict__ A,  long aStride,
    const _Float16* __restrict__ BT, long bStride,
    void* __restrict__ Cv,           long cStride,
    const float* __restrict__ bias,  float scale, int cmode) {
  const int m0B = blockIdx.y * 128;
  const int n0B = blockIdx.x * 256;
  if (cmode == 1 && n0B > m0B + 127) return;           // fully-masked block
  int kEnd = kL;
  if (cmode == 2) { int lim = m0B + 128; kEnd = lim < kL ? lim : kL; }

  const int lane = threadIdx.x & 31;
  const int wid  = threadIdx.x >> 5;
  const int wm   = wid >> 2;              // 0..1
  const int wn   = wid & 3;               // 0..3
  const int m0   = m0B + wm * 64;
  const int n0   = n0B + wn * 64;
  const int mrow = lane & 15;
  const int hi   = lane >> 4;

  A  += (long)blockIdx.z * aStride;
  BT += (long)blockIdx.z * bStride;

  const v8f vzero = {0.f, 0.f, 0.f, 0.f, 0.f, 0.f, 0.f, 0.f};
  v8f acc[4][4];
  #pragma unroll
  for (int t = 0; t < 4; ++t)
    #pragma unroll
    for (int u = 0; u < 4; ++u) acc[t][u] = vzero;

  for (int k0 = 0; k0 < kEnd; k0 += 32) {
    v16h af[4];
    #pragma unroll
    for (int t = 0; t < 4; ++t) {
      const _Float16* p = A + (long)(m0 + 16 * t + mrow) * kL + k0 + hi * 8;
      union { v16h v; v8h h[2]; } ua;
      ua.h[0] = *(const v8h*)(p);
      ua.h[1] = *(const v8h*)(p + 16);
      af[t] = ua.v;
    }
    #pragma unroll
    for (int u = 0; u < 4; ++u) {
      const _Float16* p = BT + (long)(n0 + 16 * u + mrow) * kL + k0 + hi * 8;
      union { v16h v; v8h h[2]; } ub;
      ub.h[0] = *(const v8h*)(p);
      ub.h[1] = *(const v8h*)(p + 16);
      const v16h bf = ub.v;
      #pragma unroll
      for (int t = 0; t < 4; ++t)
        acc[t][u] = __builtin_amdgcn_wmma_f32_16x16x32_f16(
            false, af[t], false, bf, (short)0, acc[t][u], false, false);
    }
  }

  // -------- epilogue --------
  #pragma unroll
  for (int t = 0; t < 4; ++t) {
    #pragma unroll
    for (int u = 0; u < 4; ++u) {
      const int col = n0 + 16 * u + mrow;
      const float bv = HAS_BIAS ? bias[col] : 0.f;
      v8f r = acc[t][u];
      #pragma unroll
      for (int j = 0; j < 8; ++j) {
        float v = r[j] * scale + bv;
        if (DO_RELU) v = v > 0.f ? v : 0.f;
        r[j] = v;
      }
      const int mrow0 = m0 + 16 * t + hi * 8;
      if (TRANS_OUT) {
        _Float16* CT = (_Float16*)Cv + (long)blockIdx.z * cStride;
        v8h hv;
        #pragma unroll
        for (int j = 0; j < 8; ++j) hv[j] = (_Float16)r[j];
        *(v8h*)(CT + (long)col * kL + mrow0) = hv;           // coalesced b128
      } else if (OUT_HALF) {
        _Float16* C = (_Float16*)Cv + (long)blockIdx.z * cStride;
        #pragma unroll
        for (int j = 0; j < 8; ++j)
          C[(long)(mrow0 + j) * kL + col] = (_Float16)r[j];
      } else {
        float* C = (float*)Cv + (long)blockIdx.z * cStride;
        #pragma unroll
        for (int j = 0; j < 8; ++j)
          C[(long)(mrow0 + j) * kL + col] = r[j];
      }
    }
  }
}

// ---------------- row softmax (1024 wide), optional causal mask ------------
template <bool CAUSAL>
__global__ __launch_bounds__(256) void softmax_kernel(
    const float* __restrict__ scores, _Float16* __restrict__ probs) {
  __shared__ float sbuf[8];
  const long row = blockIdx.x;                 // b*S + q
  const int  q   = (int)(row & (kL - 1));
  const float* src = scores + row * kL;
  _Float16*    dst = probs  + row * kL;

  float v[4];
  #pragma unroll
  for (int i = 0; i < 4; ++i) {
    const int j = threadIdx.x + i * 256;
    float x = src[j];
    if (CAUSAL && j > q) x = -__builtin_inff();
    v[i] = x;
  }
  const float m =
      block_max(fmaxf(fmaxf(v[0], v[1]), fmaxf(v[2], v[3])), sbuf);
  float e[4], s = 0.f;
  #pragma unroll
  for (int i = 0; i < 4; ++i) { e[i] = __expf(v[i] - m); s += e[i]; }  // exp(-inf)=0
  const float inv = 1.0f / block_sum(s, sbuf);
  #pragma unroll
  for (int i = 0; i < 4; ++i)
    dst[threadIdx.x + i * 256] = (_Float16)(e[i] * inv);
}

// ------- y = LN(a + resid)*gamma + beta ; optional relu; f32 + f16 out -----
// In-place safe for xout == resid (all reads complete before reductions sync).
template <bool DO_RELU>
__global__ __launch_bounds__(256) void add_ln_kernel(
    const float* __restrict__ a, const float* __restrict__ resid,
    const float* __restrict__ gamma, const float* __restrict__ beta,
    float* __restrict__ xout, _Float16* __restrict__ hout) {
  __shared__ float sbuf[8];
  const long row = blockIdx.x;
  const float* pa = a + row * kL;
  const float* pr = resid + row * kL;

  float v[4], s = 0.f, s2 = 0.f;
  #pragma unroll
  for (int i = 0; i < 4; ++i) {
    const int j = threadIdx.x + i * 256;
    const float x = pa[j] + pr[j];
    v[i] = x; s += x; s2 += x * x;
  }
  const float sum  = block_sum(s, sbuf);
  const float sum2 = block_sum(s2, sbuf);
  const float mu   = sum * (1.0f / kL);
  const float var  = sum2 * (1.0f / kL) - mu * mu;
  const float rs   = rsqrtf(var + 1e-3f);
  #pragma unroll
  for (int i = 0; i < 4; ++i) {
    const int j = threadIdx.x + i * 256;
    float y = (v[i] - mu) * rs * gamma[j] + beta[j];
    if (DO_RELU) y = y > 0.f ? y : 0.f;
    xout[row * kL + j] = y;
    if (hout) hout[row * kL + j] = (_Float16)y;
  }
}

// ---------------------------------------------------------------------------
extern "C" void kernel_launch(void* const* d_in, const int* in_sizes, int n_in,
                              void* d_out, int out_size, void* d_ws, size_t ws_size,
                              hipStream_t stream) {
  (void)in_sizes; (void)n_in; (void)out_size; (void)ws_size;
  const float* inputs  = (const float*)d_in[0];
  const float* context = (const float*)d_in[1];
  const float* Wsrc[8] = {(const float*)d_in[2], (const float*)d_in[3],
                          (const float*)d_in[4], (const float*)d_in[5],
                          (const float*)d_in[6], (const float*)d_in[7],
                          (const float*)d_in[8], (const float*)d_in[10]};
  const float* b1    = (const float*)d_in[9];
  const float* b2    = (const float*)d_in[11];
  const float* gamma = (const float*)d_in[12];
  const float* beta  = (const float*)d_in[13];

  const long LE = (long)kL * kL;        // per-batch matrix elements (1M)
  const long ME = (long)kB * LE;        // 8M elements

  // -------- workspace carve (~208 MiB) --------
  char* base = (char*)d_ws; size_t off = 0;
  auto alloc = [&](size_t bytes) -> void* {
    void* p = base + off; off += (bytes + 255) & ~(size_t)255; return p;
  };
  _Float16* xh     = (_Float16*)alloc(ME * 2);   // current x, f16
  _Float16* ch     = (_Float16*)alloc(ME * 2);   // context, f16
  _Float16* Qh     = (_Float16*)alloc(ME * 2);   // Q (later FFN hidden)
  _Float16* Kh     = (_Float16*)alloc(ME * 2);   // K
  _Float16* VTh    = (_Float16*)alloc(ME * 2);   // V transposed [b][E][S]
  float*    scores = (float*)   alloc(ME * 4);   // [b][S][S]
  _Float16* probs  = (_Float16*)alloc(ME * 2);
  float*    cbuf   = (float*)   alloc(ME * 4);   // attn / ffn output
  float*    xbuf   = (float*)   alloc(ME * 4);   // residual stream (f32)
  _Float16* WT[8];
  for (int i = 0; i < 8; ++i) WT[i] = (_Float16*)alloc(LE * 2);
  _Float16* hh = Qh;                             // FFN hidden aliases Qh

  const dim3 gg(4, 8, kB), gb(256);              // N/256, M/128, batch
  const dim3 tg(32, 32),   tb(32, 8);
  const int  rows = kB * kL;                     // 8192 row-kernels
  const float iscale = 0.03125f;                 // 1/sqrt(1024)

  // -------- precision conversion --------
  f32_to_f16_kernel<<<8192, 256, 0, stream>>>(inputs,  xh);
  f32_to_f16_kernel<<<8192, 256, 0, stream>>>(context, ch);
  for (int i = 0; i < 8; ++i)
    transpose_convert_kernel<<<tg, tb, 0, stream>>>(Wsrc[i], WT[i]);

  // -------- masked self-attention --------
  gemm_kernel<true,  false, false, false><<<gg, gb, 0, stream>>>(xh, LE, WT[0], 0, Qh,  LE, nullptr, 1.f, 0);  // Q = x@wq1
  gemm_kernel<true,  false, false, false><<<gg, gb, 0, stream>>>(xh, LE, WT[1], 0, Kh,  LE, nullptr, 1.f, 0);  // K = x@wk1
  gemm_kernel<true,  true,  false, false><<<gg, gb, 0, stream>>>(xh, LE, WT[2], 0, VTh, LE, nullptr, 1.f, 0);  // V^T
  gemm_kernel<false, false, false, false><<<gg, gb, 0, stream>>>(Qh, LE, Kh, LE, scores, LE, nullptr, iscale, 1);
  softmax_kernel<true><<<rows, 256, 0, stream>>>(scores, probs);
  gemm_kernel<false, false, false, false><<<gg, gb, 0, stream>>>(probs, LE, VTh, LE, cbuf, LE, nullptr, 1.f, 2);
  add_ln_kernel<false><<<rows, 256, 0, stream>>>(cbuf, inputs, gamma, beta, xbuf, xh);

  // -------- cross-attention (K/V from context) --------
  gemm_kernel<true,  false, false, false><<<gg, gb, 0, stream>>>(xh, LE, WT[3], 0, Qh,  LE, nullptr, 1.f, 0);  // Q = x@wq2
  gemm_kernel<true,  false, false, false><<<gg, gb, 0, stream>>>(ch, LE, WT[4], 0, Kh,  LE, nullptr, 1.f, 0);  // K = ctx@wk2
  gemm_kernel<true,  true,  false, false><<<gg, gb, 0, stream>>>(ch, LE, WT[5], 0, VTh, LE, nullptr, 1.f, 0);  // V^T
  gemm_kernel<false, false, false, false><<<gg, gb, 0, stream>>>(Qh, LE, Kh, LE, scores, LE, nullptr, iscale, 0);
  softmax_kernel<false><<<rows, 256, 0, stream>>>(scores, probs);
  gemm_kernel<false, false, false, false><<<gg, gb, 0, stream>>>(probs, LE, VTh, LE, cbuf, LE, nullptr, 1.f, 0);
  add_ln_kernel<false><<<rows, 256, 0, stream>>>(cbuf, xbuf, gamma, beta, xbuf, xh);   // in-place residual

  // -------- feed-forward --------
  gemm_kernel<true,  false, true, true ><<<gg, gb, 0, stream>>>(xh, LE, WT[6], 0, hh,   LE, b1, 1.f, 0);  // relu(x@w1+b1)
  gemm_kernel<false, false, true, false><<<gg, gb, 0, stream>>>(hh, LE, WT[7], 0, cbuf, LE, b2, 1.f, 0);  // h@w2+b2
  add_ln_kernel<true><<<rows, 256, 0, stream>>>(cbuf, xbuf, gamma, beta, (float*)d_out, nullptr);         // relu(LN(f+x))
}
